// MultiView3DDeformableAttention_68805376082096
// MI455X (gfx1250) — matrix operand order repaired and checked by reading, not compiled, not run on hardware
//
#include <hip/hip_runtime.h>
#include <hip/hip_bf16.h>
#include <math.h>

// ---------------------------------------------------------------------------
// MultiView 3D deformable attention for gfx1250 (MI455X), wave32.
// GEMMs run on v_wmma_f32_16x16x32_bf16 (fp32 -> packed bf16 at LDS staging).
// ---------------------------------------------------------------------------

typedef __bf16 bf16_t;
typedef __attribute__((ext_vector_type(16))) bf16_t v16bf;
typedef __attribute__((ext_vector_type(8)))  float  v8f;

#define TM 64
#define TN 128
#define TK 32

__device__ inline unsigned pack_bf16(float lo, float hi) {
  union { unsigned u; bf16_t h[2]; } p;
  p.h[0] = (bf16_t)lo;
  p.h[1] = (bf16_t)hi;
  return p.u;
}

__device__ inline float fast_tanh(float x) {
#if __has_builtin(__builtin_amdgcn_tanhf)
  return __builtin_amdgcn_tanhf(x);          // v_tanh_f32 (gfx1250 TRANS op)
#elif __has_builtin(__builtin_amdgcn_tanh_f32)
  return __builtin_amdgcn_tanh_f32(x);
#else
  return tanhf(x);
#endif
}

// Fragment loaders follow CDNA5 ISA 7.12.2 (wave32):
// A (16x32, 16-bit): lanes 0-15 -> M=0..15 / K halves {0..7,16..23};
//                    lanes 16-31 -> same M / K halves {8..15,24..31}.
// B (32x16, 16-bit): lanes 0-15 -> N, K {0..7,16..23}; lanes 16-31 -> N, K {8..15,24..31}.
__device__ inline v16bf load_frag_a(const bf16_t* sm, int mbase, int lane) {
  int m = mbase + (lane & 15);
  int khalf = (lane >> 4) << 3;           // 0 or 8
  v16bf f;
#pragma unroll
  for (int i = 0; i < 16; ++i) {
    int k = (i & 7) + ((i >> 3) << 4) + khalf;   // i(+8 if i>=8) + khalf
    f[i] = sm[m * TK + k];
  }
  return f;
}

__device__ inline v16bf load_frag_b(const bf16_t* sm, int nbase, int lane) {
  int n = nbase + (lane & 15);
  int kbase = (lane >> 4) << 3;           // 0 or 8
  v16bf f;
#pragma unroll
  for (int i = 0; i < 16; ++i) f[i] = sm[n * TK + kbase + (i & 7) + ((i >> 3) << 4)];
  return f;
}

// C = A(MxK,row) * B(NxK,row)^T + bias[N]; act=1 -> tanh. M%64==0, N%128==0, K%32==0.
__global__ void wmma_gemm_bias_act(const float* __restrict__ A,
                                   const float* __restrict__ B,
                                   const float* __restrict__ bias,
                                   float* __restrict__ C,
                                   int M, int N, int K, int act) {
  __shared__ unsigned As32[(TM * TK) / 2];   // bf16 pairs, row-major [TM][TK]
  __shared__ unsigned Bs32[(TN * TK) / 2];   // bf16 pairs, row-major [TN][TK]
  const bf16_t* As = (const bf16_t*)As32;
  const bf16_t* Bs = (const bf16_t*)Bs32;

  const int t    = threadIdx.x;
  const int lane = t & 31;
  const int w    = t >> 5;                 // 8 waves
  const int wm   = w >> 2;                 // 0..1  (M)
  const int wn   = w & 3;                  // 0..3  (N)
  const int tileM = blockIdx.y * TM;
  const int tileN = blockIdx.x * TN;

  v8f acc[2][2];
#pragma unroll
  for (int i = 0; i < 2; ++i)
#pragma unroll
    for (int j = 0; j < 2; ++j)
#pragma unroll
      for (int r = 0; r < 8; ++r) acc[i][j][r] = 0.0f;

  for (int k0 = 0; k0 < K; k0 += TK) {
    // Stage fp32 tiles into LDS as packed-bf16 dwords.
    // A: 1024 pairs -> 4/thread; 16 pairs per row of 32 K-cols.
#pragma unroll
    for (int e = 0; e < (TM * TK) / 512; ++e) {
      int li = e * 256 + t;
      int r = li >> 4, c = li & 15;
      float2 xv = *reinterpret_cast<const float2*>(
          A + (size_t)(tileM + r) * K + (k0 + 2 * c));
      As32[li] = pack_bf16(xv.x, xv.y);
    }
    // B: 2048 pairs -> 8/thread.
#pragma unroll
    for (int e = 0; e < (TN * TK) / 512; ++e) {
      int li = e * 256 + t;
      int r = li >> 4, c = li & 15;
      float2 xv = *reinterpret_cast<const float2*>(
          B + (size_t)(tileN + r) * K + (k0 + 2 * c));
      Bs32[li] = pack_bf16(xv.x, xv.y);
    }
    // Prefetch next K tile (global_prefetch_b8) while WMMAs run below.
    if (k0 + TK < K) {
      __builtin_prefetch(A + (size_t)(tileM + (t >> 2)) * K + (k0 + TK) + (t & 3) * 8, 0, 1);
      __builtin_prefetch(B + (size_t)(tileN + (t >> 1)) * K + (k0 + TK) + (t & 1) * 16, 0, 1);
    }
    __syncthreads();

    v16bf a0 = load_frag_a(As, wm * 32 + 0,  lane);
    v16bf a1 = load_frag_a(As, wm * 32 + 16, lane);
    v16bf b0 = load_frag_b(Bs, wn * 32 + 0,  lane);
    v16bf b1 = load_frag_b(Bs, wn * 32 + 16, lane);

    acc[0][0] = __builtin_amdgcn_wmma_f32_16x16x32_bf16(false, a0, false, b0,
                                                        (short)0, acc[0][0], false, false);
    acc[0][1] = __builtin_amdgcn_wmma_f32_16x16x32_bf16(false, a0, false, b1,
                                                        (short)0, acc[0][1], false, false);
    acc[1][0] = __builtin_amdgcn_wmma_f32_16x16x32_bf16(false, a1, false, b0,
                                                        (short)0, acc[1][0], false, false);
    acc[1][1] = __builtin_amdgcn_wmma_f32_16x16x32_bf16(false, a1, false, b1,
                                                        (short)0, acc[1][1], false, false);
    __syncthreads();
  }

  // Epilogue. C/D layout: VGPR r holds M=r (lanes 0-15) and M=r+8 (lanes 16-31); N=lane&15.
  const int mh = (lane >> 4) << 3;
#pragma unroll
  for (int i = 0; i < 2; ++i) {
#pragma unroll
    for (int j = 0; j < 2; ++j) {
      int col = tileN + wn * 32 + j * 16 + (lane & 15);
      float bv = bias ? bias[col] : 0.0f;
#pragma unroll
      for (int r = 0; r < 8; ++r) {
        int row = tileM + wm * 32 + i * 16 + r + mh;
        float vv = acc[i][j][r] + bv;
        if (act) vv = fast_tanh(vv);
        C[(size_t)row * N + col] = vv;
      }
    }
  }
}

// Softmax over contiguous rows of 384 (= NV*NL*NP*NZ). One wave32 per row.
__global__ void softmax384(float* __restrict__ attn) {
  const int lane = threadIdx.x & 31;
  const int w    = threadIdx.x >> 5;
  const long row = (long)blockIdx.x * 8 + w;
  float* p = attn + row * 384;
  float vals[12];
  float m = -3.402823e38f;
#pragma unroll
  for (int i = 0; i < 12; ++i) { vals[i] = p[lane + i * 32]; m = fmaxf(m, vals[i]); }
#pragma unroll
  for (int s = 16; s >= 1; s >>= 1) m = fmaxf(m, __shfl_xor(m, s, 32));
  float sum = 0.0f;
#pragma unroll
  for (int i = 0; i < 12; ++i) { vals[i] = __expf(vals[i] - m); sum += vals[i]; }
#pragma unroll
  for (int s = 16; s >= 1; s >>= 1) sum += __shfl_xor(sum, s, 32);
  float inv = 1.0f / sum;
#pragma unroll
  for (int i = 0; i < 12; ++i) p[lane + i * 32] = vals[i] * inv;
}

__device__ inline int iclamp(int x, int lo, int hi) {
  return x < lo ? lo : (x > hi ? hi : x);
}

// Deformable bilinear sampling. One wave32 per (q,h); lane == channel d (HD=32).
// valp layout: [v*5440 + s][256] with emb col = h*32 + d (GEMM output, row-major).
__global__ void deform_sample(const float* __restrict__ ref,
                              const float* __restrict__ offs,
                              const float* __restrict__ attn,
                              const float* __restrict__ valp,
                              float* __restrict__ out) {
  const int lane = threadIdx.x & 31;
  const int w    = threadIdx.x >> 5;
  const int gw   = blockIdx.x * 8 + w;     // 0..12799
  const int q    = gw >> 3;
  const int h    = gw & 7;

  const int Hs[4]    = {64, 32, 16, 8};
  const int Ws[4]    = {64, 32, 16, 8};
  const int start[4] = {0, 4096, 5120, 5376};

  const float* offq = offs + (size_t)q * 6144;
  const float* attq = attn + (size_t)q * 3072;
  const float* refq = ref  + (size_t)q * (6 * 4 * 4 * 2);

  float acc = 0.0f;
  for (int v = 0; v < 6; ++v) {
    for (int l = 0; l < 4; ++l) {
      const int Wl = Ws[l], Hl = Hs[l];
      const float sx = 0.5f * (float)(Wl - 1);
      const float sy = 0.5f * (float)(Hl - 1);
      const float* fm = valp + ((size_t)(v * 5440 + start[l])) * 256 + h * 32 + lane;
      for (int z = 0; z < 4; ++z) {
        const float rx = refq[(((v * 4 + l) * 4 + z) * 2) + 0];
        const float ry = refq[(((v * 4 + l) * 4 + z) * 2) + 1];
        for (int p = 0; p < 4; ++p) {
          const int a = (((h * 6 + v) * 4 + l) * 4 + p) * 4 + z;
          const float gx = fminf(fmaxf(rx + offq[a * 2 + 0], -1.0f), 1.0f);
          const float gy = fminf(fmaxf(ry + offq[a * 2 + 1], -1.0f), 1.0f);
          const float wgt = attq[a];
          const float x = (gx + 1.0f) * sx;
          const float y = (gy + 1.0f) * sy;
          const float x0f = floorf(x), y0f = floorf(y);
          const float wx = x - x0f, wy = y - y0f;
          const int x0 = iclamp((int)x0f, 0, Wl - 1);
          const int y0 = iclamp((int)y0f, 0, Hl - 1);
          const int x1 = iclamp(x0 + 1, 0, Wl - 1);
          const int y1 = iclamp(y0 + 1, 0, Hl - 1);
          const float v00 = fm[(size_t)(y0 * Wl + x0) * 256];
          const float v01 = fm[(size_t)(y0 * Wl + x1) * 256];
          const float v10 = fm[(size_t)(y1 * Wl + x0) * 256];
          const float v11 = fm[(size_t)(y1 * Wl + x1) * 256];
          const float s = (1.0f - wy) * ((1.0f - wx) * v00 + wx * v01) +
                          wy          * ((1.0f - wx) * v10 + wx * v11);
          acc += wgt * s;
        }
      }
    }
  }
  out[(size_t)q * 256 + h * 32 + lane] = acc;
}

extern "C" void kernel_launch(void* const* d_in, const int* in_sizes, int n_in,
                              void* d_out, int out_size, void* d_ws, size_t ws_size,
                              hipStream_t stream) {
  (void)in_sizes; (void)n_in; (void)out_size; (void)ws_size;
  const float* queries = (const float*)d_in[0];
  const float* refpts  = (const float*)d_in[1];
  const float* value   = (const float*)d_in[2];
  // d_in[3] = value_spatial_shapes (compile-time constants here)
  const float* Wv      = (const float*)d_in[4];
  const float* Woff    = (const float*)d_in[5];
  const float* boff    = (const float*)d_in[6];
  const float* Wattn   = (const float*)d_in[7];
  const float* battn   = (const float*)d_in[8];
  const float* Wout    = (const float*)d_in[9];
  float* out = (float*)d_out;

  char* ws = (char*)d_ws;
  float* offs = (float*)(ws);                         // 1600*6144 f32 = 39321600 B
  float* attn = (float*)(ws + 39321600);              // 1600*3072 f32 = 19660800 B
  float* valp = (float*)(ws + 58982400);              // 32640*256 f32 = 33423360 B
  float* samp = (float*)(ws + 92405760);              // 1600*256  f32 =  1638400 B

  const dim3 blk(256);
  // offsets: tanh(Q @ Woff^T + boff)
  wmma_gemm_bias_act<<<dim3(6144 / TN, 1600 / TM), blk, 0, stream>>>(
      queries, Woff, boff, offs, 1600, 6144, 256, 1);
  // attention logits: Q @ Wattn^T + battn
  wmma_gemm_bias_act<<<dim3(3072 / TN, 1600 / TM), blk, 0, stream>>>(
      queries, Wattn, battn, attn, 1600, 3072, 256, 0);
  // value projection: value @ Wv^T
  wmma_gemm_bias_act<<<dim3(256 / TN, 32640 / TM), blk, 0, stream>>>(
      value, Wv, nullptr, valp, 32640, 256, 256, 0);
  // softmax over 384-wide groups (12800 rows, 8 waves/block)
  softmax384<<<1600, 256, 0, stream>>>(attn);
  // deformable bilinear sampling + attention-weighted accumulation
  deform_sample<<<1600, 256, 0, stream>>>(refpts, offs, attn, valp, samp);
  // output projection: samp @ Wout^T
  wmma_gemm_bias_act<<<dim3(256 / TN, 1600 / TM), blk, 0, stream>>>(
      samp, Wout, nullptr, out, 1600, 256, 256, 0);
}
